// LSTMLanguageModel_3478923510556
// MI455X (gfx1250) — matrix operand order repaired and checked by reading, not compiled, not run on hardware
//
#include <hip/hip_runtime.h>

#define VOCAB  32000
#define HIDDEN 1024
#define BATCH  8
#define SEQ    512
#define ROWS   (BATCH * SEQ)   // 4096
#define GATES  (4 * HIDDEN)    // 4096

typedef __attribute__((ext_vector_type(16))) __bf16       v16bf;
typedef __attribute__((ext_vector_type(8)))  float        v8f;
typedef __attribute__((ext_vector_type(8)))  unsigned int v8ui;
typedef __attribute__((ext_vector_type(4)))  unsigned int v4ui;

struct U32x8 { uint4 a, b; };   // 32B, bit_cast source for v16bf fragments

__device__ __forceinline__ unsigned short f2bf(float f) {
  unsigned int u = __float_as_uint(f);
  u += 0x7FFFu + ((u >> 16) & 1u);          // round-to-nearest-even
  return (unsigned short)(u >> 16);
}

// ---------------- prep: transpose f32 [K][N] -> bf16 [N][K] ----------------
__global__ __launch_bounds__(256) void transpose_to_bf16(
    const float* __restrict__ in, unsigned short* __restrict__ out, int N, int K) {
  __shared__ float tile[32][33];
  const int tx = threadIdx.x, ty = threadIdx.y;       // 32 x 8
  const int n0 = blockIdx.x * 32, k0 = blockIdx.y * 32;
  #pragma unroll
  for (int i = 0; i < 4; ++i)
    tile[ty + i * 8][tx] = in[(size_t)(k0 + ty + i * 8) * N + n0 + tx];
  __syncthreads();
  #pragma unroll
  for (int i = 0; i < 4; ++i)
    out[(size_t)(n0 + ty + i * 8) * K + k0 + tx] = f2bf(tile[tx][ty + i * 8]);
}

// ---------------- prep: weight-tied embedding gather (writes bf16) ----------------
__global__ void embed_gather(const int* __restrict__ ids,
                             const float* __restrict__ Wout,
                             unsigned short* __restrict__ emb_bf) {
  int r = blockIdx.y;
  int h = blockIdx.x * 256 + threadIdx.x;
  int id = ids[r];
  emb_bf[(size_t)r * HIDDEN + h] = f2bf(Wout[(size_t)h * VOCAB + id]);
}

// ---------------- TDM 2D tile load: global bf16 [rows x K] -> padded LDS ----------------
// LDS row layout: 32 bf16 (16 dwords) + 4 pad dwords = 20 dwords (bank-conflict-free)
__device__ __forceinline__ void tdm_load_2d(unsigned lds_off, const void* gptr,
                                            unsigned dim0 /*K*/, unsigned dim1 /*rows*/,
                                            unsigned tile1 /*rows in tile*/) {
  unsigned long long ga = (unsigned long long)gptr;
  v4ui g0;
  g0[0] = 1u;                                        // count=1, user descriptor
  g0[1] = lds_off;                                   // LDS byte address
  g0[2] = (unsigned)ga;                              // global_addr[31:0]
  g0[3] = ((unsigned)(ga >> 32) & 0x01FFFFFFu) | (2u << 30);  // addr[56:32] | type=2
  v8ui g1;
  g1[0] = (1u << 16)        // data_size = 2 bytes (bf16)
        | (1u << 20)        // pad_enable
        | (3u << 22)        // pad_interval: 2^(3+1)=16 dwords of data
        | (3u << 25);       // pad_amount: 3+1 = 4 dwords of pad
  g1[1] = (dim0 & 0xFFFFu) << 16;                    // tensor_dim0[15:0]  @ bits 63:48
  g1[2] = (dim0 >> 16) | ((dim1 & 0xFFFFu) << 16);   // dim0[31:16], dim1[15:0]
  g1[3] = (dim1 >> 16) | (32u << 16);                // dim1[31:16], tile_dim0=32
  g1[4] = tile1;                                     // tile_dim1 (tile_dim2 = 0)
  g1[5] = dim0;                                      // tensor_dim0_stride[31:0] = K
  g1[6] = 0u;
  g1[7] = 0u;
  v4ui gz = {0u, 0u, 0u, 0u};                        // groups 2/3: dims 2..4 unused
  asm volatile("tensor_load_to_lds %0, %1, %2, %3"
               :: "s"(g0), "s"(g1), "s"(gz), "s"(gz)
               : "memory");
}

// ---------------- C = A(bf16) * Bt(bf16)^T + bias, f32 accum via WMMA ----------------
#define BM 128
#define BN 128
#define BK 32
#define LDR 20                         // dwords per LDS row (16 data + 4 pad)
#define BUF_DW   ((BM + BN) * LDR)     // 5120 dwords per buffer
#define BUF_BYTE (BUF_DW * 4)          // 20480 B
#define B_OFF_BYTE (BM * LDR * 4)      // 10240 B: B tile offset within a buffer

__global__ __launch_bounds__(256) void gemm_bias_bf16_tdm(
    const unsigned short* __restrict__ A,    // [M,K] bf16 row-major
    const unsigned short* __restrict__ Bt,   // [N,K] bf16 row-major (pre-transposed)
    const float* __restrict__ bias,          // [N]
    float* __restrict__ C,                   // [M,N] f32
    int M, int N, int K) {
  extern __shared__ unsigned int lds[];      // 2 x (A tile + B tile), padded rows

  const int tid  = threadIdx.x;
  const int lane = tid & 31;
  const int wave = tid >> 5;                 // 0..7
  const int m0 = blockIdx.y * BM;
  const int n0 = blockIdx.x * BN;
  const int wm = wave & 3;                   // 4 m-groups of 32 rows (2 subtiles)
  const int wn = wave >> 2;                  // 2 n-groups of 64 cols (4 subtiles)
  const int l15 = lane & 15;
  const int hi  = lane >> 4;                 // lane half per ISA fragment layout

  int nCol[4];
  v8f acc[2][4];
  #pragma unroll
  for (int s = 0; s < 4; ++s) {
    nCol[s] = n0 + (wn * 4 + s) * 16 + l15;
    float bv = bias[nCol[s]];
    #pragma unroll
    for (int j = 0; j < 2; ++j)
      #pragma unroll
      for (int r = 0; r < 8; ++r) acc[j][s][r] = bv;
  }

  const int NS = K / BK;                     // pipeline stages
  if (tid == 0) {                            // TDM ignores EXEC; wave 0 issues DMAs
    tdm_load_2d(0 * BUF_BYTE,              A  + (size_t)m0 * K + 0,  (unsigned)K, (unsigned)M, BM);
    tdm_load_2d(0 * BUF_BYTE + B_OFF_BYTE, Bt + (size_t)n0 * K + 0,  (unsigned)K, (unsigned)N, BN);
    tdm_load_2d(1 * BUF_BYTE,              A  + (size_t)m0 * K + BK, (unsigned)K, (unsigned)M, BM);
    tdm_load_2d(1 * BUF_BYTE + B_OFF_BYTE, Bt + (size_t)n0 * K + BK, (unsigned)K, (unsigned)N, BN);
  }

  for (int s = 0; s < NS; ++s) {
    const int cur = s & 1;
    if (tid == 0) {
      if (s + 1 < NS) __builtin_amdgcn_s_wait_tensorcnt(2);  // oldest A+B pair landed,
      else            __builtin_amdgcn_s_wait_tensorcnt(0);  // next pair stays in flight
    }
    __syncthreads();

    const unsigned int* bufA = lds + cur * BUF_DW;
    const unsigned int* bufB = bufA + BM * LDR;

    // A fragments (16x32 bf16): lane half 0 -> K{0..7,16..23}, half 1 -> +8
    const int oa = hi * 4;
    v16bf af[2];
    #pragma unroll
    for (int j = 0; j < 2; ++j) {
      const unsigned int* rA = &bufA[((wm * 2 + j) * 16 + l15) * LDR];
      U32x8 t;
      t.a = *(const uint4*)(rA + oa);
      t.b = *(const uint4*)(rA + oa + 8);
      af[j] = __builtin_bit_cast(v16bf, t);
    }

    // B fragments (32x16 bf16): lane half 0 -> K{0..7,16..23}, half 1 -> +8
    const int ob = hi * 4;
    #pragma unroll
    for (int sb = 0; sb < 4; ++sb) {
      const unsigned int* rB = &bufB[((wn * 4 + sb) * 16 + l15) * LDR];
      U32x8 t;
      t.a = *(const uint4*)(rB + ob);
      t.b = *(const uint4*)(rB + ob + 8);
      v16bf bfr = __builtin_bit_cast(v16bf, t);
      #pragma unroll
      for (int j = 0; j < 2; ++j)
        acc[j][sb] = __builtin_amdgcn_wmma_f32_16x16x32_bf16(false, af[j], false, bfr,
                                                             (short)0, acc[j][sb], false, false);
    }
    __syncthreads();                         // all reads of buf[cur] done

    if (tid == 0 && s + 2 < NS) {            // refill freed buffer for stage s+2
      int kb = (s + 2) * BK;
      tdm_load_2d(cur * BUF_BYTE,              A  + (size_t)m0 * K + kb, (unsigned)K, (unsigned)M, BM);
      tdm_load_2d(cur * BUF_BYTE + B_OFF_BYTE, Bt + (size_t)n0 * K + kb, (unsigned)K, (unsigned)N, BN);
    }
  }

  #pragma unroll
  for (int j = 0; j < 2; ++j)
    #pragma unroll
    for (int s = 0; s < 4; ++s)
      #pragma unroll
      for (int r = 0; r < 8; ++r) {
        int m = m0 + (wm * 2 + j) * 16 + r + hi * 8;  // C/D layout: lanes 16..31 hold M+8
        __builtin_nontemporal_store(acc[j][s][r], &C[(size_t)m * N + nCol[s]]);
      }
}

// ---------------- one LSTM time step (fp32; M=8, latency-bound) ----------------
__global__ __launch_bounds__(512) void lstm_step(
    const float* __restrict__ xg,       // [B,S,4H]
    const float* __restrict__ Wh,       // [H,4H]
    float* __restrict__ hs,             // [B,S,H] f32 (reads t-1, writes t)
    unsigned short* __restrict__ hsb,   // [B,S,H] bf16 copy for the output GEMM
    float* __restrict__ cbuf,           // [B,H]
    int t) {
  __shared__ float sh[BATCH * HIDDEN];
  const int tid = threadIdx.y * 64 + threadIdx.x;
  if (t > 0) {
    for (int idx = tid; idx < BATCH * HIDDEN; idx += 512) {
      int b = idx >> 10, k = idx & (HIDDEN - 1);
      sh[idx] = hs[(size_t)(b * SEQ + (t - 1)) * HIDDEN + k];
    }
  } else {
    for (int idx = tid; idx < BATCH * HIDDEN; idx += 512) sh[idx] = 0.f;
  }
  __syncthreads();

  const int j = blockIdx.x * 64 + threadIdx.x;
  const int b = threadIdx.y;
  float ai = 0.f, af_ = 0.f, ag = 0.f, ao = 0.f;
  const float* hb = &sh[b * HIDDEN];
  const float* wj = Wh + j;
  #pragma unroll 4
  for (int k = 0; k < HIDDEN; ++k) {
    float hk = hb[k];
    const float* wr = wj + (size_t)k * GATES;
    ai  += hk * wr[0];
    af_ += hk * wr[HIDDEN];
    ag  += hk * wr[2 * HIDDEN];
    ao  += hk * wr[3 * HIDDEN];
  }
  const float* x = xg + (size_t)(b * SEQ + t) * GATES + j;
  ai += x[0]; af_ += x[HIDDEN]; ag += x[2 * HIDDEN]; ao += x[3 * HIDDEN];

  float ii = 1.f / (1.f + expf(-ai));
  float ff = 1.f / (1.f + expf(-af_));
  float gg = tanhf(ag);
  float oo = 1.f / (1.f + expf(-ao));
  float cp = (t == 0) ? 0.f : cbuf[b * HIDDEN + j];
  float cn = ff * cp + ii * gg;
  cbuf[b * HIDDEN + j] = cn;
  float hn = oo * tanhf(cn);
  size_t o = (size_t)(b * SEQ + t) * HIDDEN + j;
  hs[o]  = hn;
  hsb[o] = f2bf(hn);
}

// ---------------- host launcher ----------------
extern "C" void kernel_launch(void* const* d_in, const int* in_sizes, int n_in,
                              void* d_out, int out_size, void* d_ws, size_t ws_size,
                              hipStream_t stream) {
  const int*   ids    = (const int*)d_in[0];
  const float* W_out  = (const float*)d_in[1];
  const float* b_out  = (const float*)d_in[2];
  const float* Wx     = (const float*)d_in[3];
  const float* Wh     = (const float*)d_in[4];
  const float* b_lstm = (const float*)d_in[5];
  float* out = (float*)d_out;

  // workspace layout
  float* xg = (float*)d_ws;                              // 8*512*4096 f32 (64 MB)
  float* hs = xg + (size_t)ROWS * GATES;                 // 4096*1024 f32 (16 MB)
  float* cb = hs + (size_t)ROWS * HIDDEN;                // 8*1024 f32
  unsigned short* emb_bf = (unsigned short*)(cb + BATCH * HIDDEN);  // 4096*1024 bf16
  unsigned short* hsb    = emb_bf + (size_t)ROWS * HIDDEN;          // 4096*1024 bf16
  unsigned short* wxt    = hsb    + (size_t)ROWS * HIDDEN;          // [4096][1024] bf16
  unsigned short* woutt  = wxt    + (size_t)GATES * HIDDEN;         // [32000][1024] bf16

  // prep: convert + transpose weights once; gather embeddings as bf16
  transpose_to_bf16<<<dim3(GATES / 32, HIDDEN / 32), dim3(32, 8), 0, stream>>>(
      Wx, wxt, GATES, HIDDEN);
  transpose_to_bf16<<<dim3(VOCAB / 32, HIDDEN / 32), dim3(32, 8), 0, stream>>>(
      W_out, woutt, VOCAB, HIDDEN);
  embed_gather<<<dim3(HIDDEN / 256, ROWS), 256, 0, stream>>>(ids, W_out, emb_bf);

  const size_t ldsBytes = 2 * (size_t)BUF_BYTE;          // 40960 B (double-buffered)

  // x_gates = emb @ Wx + b_lstm   (M=4096, N=4096, K=1024)
  gemm_bias_bf16_tdm<<<dim3(GATES / BN, ROWS / BM), 256, ldsBytes, stream>>>(
      emb_bf, wxt, b_lstm, xg, ROWS, GATES, HIDDEN);

  // recurrence: 512 dependent steps, fp32
  for (int t = 0; t < SEQ; ++t)
    lstm_step<<<dim3(HIDDEN / 64), dim3(64, 8), 0, stream>>>(xg, Wh, hs, hsb, cb, t);

  // logits = hs @ W_out + b_out   (M=4096, N=32000, K=1024)
  gemm_bias_bf16_tdm<<<dim3(VOCAB / BN, ROWS / BM), 256, ldsBytes, stream>>>(
      hsb, woutt, b_out, out, ROWS, VOCAB, HIDDEN);
}
